// SimpleAttentionWithKeyValueCache_51548197487409
// MI455X (gfx1250) — matrix operand order repaired and checked by reading, not compiled, not run on hardware
//
#include <hip/hip_runtime.h>
#include <cstdint>
#include <cstddef>

// ---------------------------------------------------------------------------
// SimpleAttentionWithKeyValueCache for MI455X (gfx1250, wave32, WMMA)
// B=8, Q_LEN=16, E=2048, CACHE_LEN=4096, past_kv_len=4080
// Memory-bound (~563 MB @ 23.3 TB/s); all GEMMs via v_wmma_f32_16x16x32_bf16.
// f32->bf16 conversions via __builtin_convertvector / (__bf16) casts, which
// lower to v_cvt_pk_bf16_f32 on gfx1250.
// ---------------------------------------------------------------------------

#define EDIM  2048
#define CLEN  4096
#define NB    8
#define QL    16
#define PAST  4080
#define SCALE 0.022097086912079608f   // 1/sqrt(2048)

typedef __attribute__((ext_vector_type(16))) __bf16         v16bf;
typedef __attribute__((ext_vector_type(16))) float          v16f;
typedef __attribute__((ext_vector_type(8)))  float          v8f;
typedef __attribute__((ext_vector_type(8)))  unsigned int   v8u;
typedef __attribute__((ext_vector_type(8)))  unsigned short v8us;

// ---- helpers --------------------------------------------------------------
__device__ __forceinline__ unsigned short f2bf(float a) {
    return __builtin_bit_cast(unsigned short, (__bf16)a);
}
__device__ __forceinline__ v16bf mk16(uint4 a, uint4 b) {
    v8u r;
    r[0] = a.x; r[1] = a.y; r[2] = a.z; r[3] = a.w;
    r[4] = b.x; r[5] = b.y; r[6] = b.z; r[7] = b.w;
    return __builtin_bit_cast(v16bf, r);
}
// 16 contiguous f32 (64B aligned) -> 16 bf16 (B-operand slice)
__device__ __forceinline__ v16bf cvt16(const float* __restrict__ p) {
    v16f v = *(const v16f*)p;
    return __builtin_convertvector(v, v16bf);
}
// A-operand: 8 f32 at p0 (K = h*16..+7) and 8 f32 at p1 (K = h*16+8..+15)
__device__ __forceinline__ v16bf cvtA(const float* __restrict__ p0,
                                      const float* __restrict__ p1) {
    v8f a = *(const v8f*)p0;
    v8f b = *(const v8f*)p1;
    v16f cat = __builtin_shufflevector(a, b, 0, 1, 2, 3, 4, 5, 6, 7,
                                             8, 9, 10, 11, 12, 13, 14, 15);
    return __builtin_convertvector(cat, v16bf);
}
// 8 bf16 at p0 and 8 bf16 at p1 (each 16B aligned)
__device__ __forceinline__ v16bf ld16bf(const unsigned short* __restrict__ p0,
                                        const unsigned short* __restrict__ p1) {
    uint4 a = *(const uint4*)p0;
    uint4 b = *(const uint4*)p1;
    return mk16(a, b);
}

// ---------------------------------------------------------------------------
// Kernel 1: QKV projection.  D[b,q,f] = sum_e x[b,q,e]*W[f,e] + bias[f]
// grid = (16 f-tiles of 128, 3 matrices, 8 batches); 8 waves/block,
// each wave computes one 16(q) x 16(f) tile, looping e in steps of 32.
// ---------------------------------------------------------------------------
__global__ __launch_bounds__(256)
void qkv_kernel(const float* __restrict__ x,
                const float* __restrict__ Wq, const float* __restrict__ bq,
                const float* __restrict__ Wk, const float* __restrict__ bk,
                const float* __restrict__ Wv, const float* __restrict__ bv,
                unsigned short* __restrict__ Qb,
                unsigned short* __restrict__ Kb,
                unsigned short* __restrict__ Vb)
{
    const int b = blockIdx.z, m = blockIdx.y;
    const float* W    = (m == 0) ? Wq : (m == 1) ? Wk : Wv;
    const float* bias = (m == 0) ? bq : (m == 1) ? bk : bv;
    unsigned short* dst = (m == 0) ? Qb : (m == 1) ? Kb : Vb;

    const int wave = threadIdx.x >> 5, lane = threadIdx.x & 31;
    const int n = lane & 15, h = lane >> 4;          // col / K-half select
    const int f = blockIdx.x * 128 + wave * 16 + n;  // output column (B op)
    const float* xrow = x + (size_t)(b * QL + n) * EDIM;  // A row M = lane&15
    const float* wrow = W + (size_t)f * EDIM;

    v8f acc = {};
#pragma unroll 2
    for (int e0 = 0; e0 < EDIM; e0 += 32) {
        v16bf A  = cvtA(xrow + e0 + h * 16, xrow + e0 + h * 16 + 8);
        v16bf Bm = cvt16(wrow + e0 + h * 16);
        acc = __builtin_amdgcn_wmma_f32_16x16x32_bf16(false, A, false, Bm,
                                                      (short)0, acc, false, false);
    }
    const float bv_ = bias[f];
#pragma unroll
    for (int r = 0; r < 8; ++r) {
        const int q = r + 8 * h;
        dst[(size_t)(b * QL + q) * EDIM + f] = f2bf(acc[r] + bv_);
    }
}

// ---------------------------------------------------------------------------
// Kernel 2: scores.  S[b,q,k] = (sum_e Q[b,q,e]*K[b,k,e]) * scale + mask
// K rows < 4080 stream from fp32 k_cache; rows >= 4080 from projected Kb.
// grid = (32 k-chunks of 128, 8 batches); wave tile = 16(q) x 16(k).
// ---------------------------------------------------------------------------
__global__ __launch_bounds__(256)
void score_kernel(const float* __restrict__ mask,
                  const float* __restrict__ kc,
                  const unsigned short* __restrict__ Qb,
                  const unsigned short* __restrict__ Kb,
                  float* __restrict__ S)
{
    const int b = blockIdx.y;
    const int wave = threadIdx.x >> 5, lane = threadIdx.x & 31;
    const int n = lane & 15, h = lane >> 4;
    const int kbase = blockIdx.x * 128 + wave * 16;
    const int krow  = kbase + n;
    const bool isnew = (kbase >= PAST);        // 4080 is 16-aligned: tile-uniform

    const unsigned short* qrow = Qb + (size_t)(b * QL + n) * EDIM;
    const float* kcr = kc + ((size_t)b * CLEN + krow) * EDIM;
    const unsigned short* knr =
        Kb + (size_t)(b * QL + (isnew ? (krow - PAST) : 0)) * EDIM;

    v8f acc = {};
#pragma unroll 2
    for (int e0 = 0; e0 < EDIM; e0 += 32) {
        v16bf A = ld16bf(qrow + e0 + h * 16, qrow + e0 + h * 16 + 8);
        v16bf Bm;
        if (isnew) {
            Bm = ld16bf(knr + e0 + h * 16, knr + e0 + h * 16 + 8);
        } else {
            __builtin_prefetch(kcr + e0 + 256, 0, 1);   // global_prefetch_b8
            Bm = cvt16(kcr + e0 + h * 16);
        }
        acc = __builtin_amdgcn_wmma_f32_16x16x32_bf16(false, A, false, Bm,
                                                      (short)0, acc, false, false);
    }
#pragma unroll
    for (int r = 0; r < 8; ++r) {
        const int q = r + 8 * h;
        const size_t idx = (size_t)(b * QL + q) * CLEN + krow;
        S[idx] = acc[r] * SCALE + mask[idx];
    }
}

// ---------------------------------------------------------------------------
// Kernel 3: row softmax over 4096, probs written as bf16.
// grid = 128 rows (b*16+q), 256 threads.
// ---------------------------------------------------------------------------
__global__ __launch_bounds__(256)
void softmax_kernel(const float* __restrict__ S, unsigned short* __restrict__ P)
{
    __shared__ float red[256];
    const int row = blockIdx.x, tid = threadIdx.x;
    const float* s = S + (size_t)row * CLEN;

    float m = -3.4e38f;
    for (int i = tid; i < CLEN; i += 256) m = fmaxf(m, s[i]);
    red[tid] = m; __syncthreads();
    for (int st = 128; st > 0; st >>= 1) {
        if (tid < st) red[tid] = fmaxf(red[tid], red[tid + st]);
        __syncthreads();
    }
    const float mx = red[0]; __syncthreads();

    float sum = 0.f;
    for (int i = tid; i < CLEN; i += 256) sum += __expf(s[i] - mx);
    red[tid] = sum; __syncthreads();
    for (int st = 128; st > 0; st >>= 1) {
        if (tid < st) red[tid] += red[tid + st];
        __syncthreads();
    }
    const float inv = 1.0f / red[0];

    unsigned short* p = P + (size_t)row * CLEN;
    for (int i = tid; i < CLEN; i += 256) p[i] = f2bf(__expf(s[i] - mx) * inv);
}

// ---------------------------------------------------------------------------
// Kernel 4: O[b,q,f] = sum_k P[b,q,k] * V[b,k,f].
// B-operand needs V columns along k -> stage 32x128 V tile transposed in LDS
// ([f][k], row stride 40 halves; conflict-free b128 reads).  Each staging
// thread owns a k-PAIR so adjacent (k,k+1) bf16 values pack into one
// ds_store_b32 (pairs never straddle the 4080 old/new boundary).  Global
// loads for tile i+1 are double-buffered in registers to overlap the WMMA
// of tile i.  grid = (16 f-tiles of 128, 8 batches); 8 waves/block.
// ---------------------------------------------------------------------------
struct StageReg {
    v8f   fa, fc;   // f32 path (old cache rows)
    uint4 ua, uc;   // bf16 path (new projected rows)
};

__global__ __launch_bounds__(256)
void pv_kernel(const float* __restrict__ vc,
               const unsigned short* __restrict__ Vb,
               const unsigned short* __restrict__ P,
               float* __restrict__ out)
{
    __shared__ __align__(16) unsigned short ldsv[128 * 40];  // [f_local][k_local]
    unsigned int* ldsd = (unsigned int*)ldsv;                // dword view, stride 20
    const int b = blockIdx.y;
    const int fblk = blockIdx.x * 128;
    const int tid = threadIdx.x;
    const int wave = tid >> 5, lane = tid & 31, n = lane & 15, h = lane >> 4;
    const int kp = tid >> 4;            // staging: k-pair 0..15 (rows 2kp, 2kp+1)
    const int fo = (tid & 15) * 8;      // staging: 8 consecutive f

    const unsigned short* prow = P + (size_t)(b * QL + n) * CLEN;
    v8f acc = {};

    StageReg cur, nxt;
    {   // prime tile 0 (k rows 0..31 are always old cache rows)
        const float* s0 = vc + ((size_t)b * CLEN + 2 * kp) * EDIM + fblk + fo;
        cur.fa = *(const v8f*)s0;
        cur.fc = *(const v8f*)(s0 + EDIM);
    }

    for (int k0 = 0; k0 < CLEN; k0 += 32) {
        const bool oldrow = (k0 + 2 * kp) < PAST;
        __syncthreads();                 // previous compute done reading LDS
        if (oldrow) {
#pragma unroll
            for (int j = 0; j < 8; ++j) {
                const unsigned int d = (unsigned int)f2bf(cur.fa[j]) |
                                       ((unsigned int)f2bf(cur.fc[j]) << 16);
                ldsd[(fo + j) * 20 + kp] = d;
            }
        } else {
            v8us a = __builtin_bit_cast(v8us, cur.ua);
            v8us c = __builtin_bit_cast(v8us, cur.uc);
#pragma unroll
            for (int j = 0; j < 8; ++j) {
                const unsigned int d = (unsigned int)a[j] |
                                       ((unsigned int)c[j] << 16);
                ldsd[(fo + j) * 20 + kp] = d;
            }
        }
        __syncthreads();

        // issue next tile's global loads; consumed at next iteration's store
        if (k0 + 32 < CLEN) {
            const int kg = k0 + 32 + 2 * kp;
            if (kg < PAST) {
                const float* s0 = vc + ((size_t)b * CLEN + kg) * EDIM + fblk + fo;
                nxt.fa = *(const v8f*)s0;
                nxt.fc = *(const v8f*)(s0 + EDIM);
            } else {
                const unsigned short* s0 =
                    Vb + (size_t)(b * QL + (kg - PAST)) * EDIM + fblk + fo;
                nxt.ua = *(const uint4*)s0;
                nxt.uc = *(const uint4*)(s0 + EDIM);
            }
        }

        v16bf A = ld16bf(prow + k0 + h * 16, prow + k0 + h * 16 + 8);
        const unsigned short* lp = &ldsv[(wave * 16 + n) * 40 + h * 16];
        v16bf Bm = ld16bf(lp, lp + 8);
        acc = __builtin_amdgcn_wmma_f32_16x16x32_bf16(false, A, false, Bm,
                                                      (short)0, acc, false, false);
        cur = nxt;
    }

    const int f = fblk + wave * 16 + n;
#pragma unroll
    for (int r = 0; r < 8; ++r)
        out[(size_t)(b * QL + (r + 8 * h)) * EDIM + f] = acc[r];
}

// ---------------------------------------------------------------------------
extern "C" void kernel_launch(void* const* d_in, const int* in_sizes, int n_in,
                              void* d_out, int out_size, void* d_ws, size_t ws_size,
                              hipStream_t stream)
{
    const float* x    = (const float*)d_in[0];
    const float* mask = (const float*)d_in[1];
    const float* kc   = (const float*)d_in[2];
    const float* vc   = (const float*)d_in[3];
    const float* Wq   = (const float*)d_in[4];
    const float* bq   = (const float*)d_in[5];
    const float* Wk   = (const float*)d_in[6];
    const float* bk   = (const float*)d_in[7];
    const float* Wv   = (const float*)d_in[8];
    const float* bv   = (const float*)d_in[9];
    float* out = (float*)d_out;

    char* ws = (char*)d_ws;
    unsigned short* Qb = (unsigned short*)(ws + 0);        // 8*16*2048 bf16
    unsigned short* Kb = (unsigned short*)(ws + 524288);   // 8*16*2048 bf16
    unsigned short* Vb = (unsigned short*)(ws + 1048576);  // 8*16*2048 bf16
    float*          S  = (float*)(ws + 1572864);           // 8*16*4096 f32
    unsigned short* P  = (unsigned short*)(ws + 3670016);  // 8*16*4096 bf16
    // total workspace use: 4,718,592 bytes

    qkv_kernel<<<dim3(16, 3, NB), 256, 0, stream>>>(x, Wq, bq, Wk, bk, Wv, bv,
                                                    Qb, Kb, Vb);
    score_kernel<<<dim3(32, NB), 256, 0, stream>>>(mask, kc, Qb, Kb, S);
    softmax_kernel<<<dim3(NB * QL), 256, 0, stream>>>(S, P);
    pv_kernel<<<dim3(16, NB), 256, 0, stream>>>(vc, Vb, P, out);
}
